// Decoder_9706626089598
// MI455X (gfx1250) — matrix operand order repaired and checked by reading, not compiled, not run on hardware
//
#include <hip/hip_runtime.h>

typedef __attribute__((ext_vector_type(16))) _Float16 v16h;
typedef __attribute__((ext_vector_type(8)))  _Float16 v8h;
typedef __attribute__((ext_vector_type(8)))  float    v8f;

#define B_TOT   512
#define HDIM    128
#define G4      512     // 4*H
#define LATD    16
#define OUTD    12
#define BT      16      // batch rows per workgroup
#define WPITCH  136     // halves per W_hh row in LDS (pad 128 -> 136 to spread banks)
#define HPITCH  136     // halves per h row in LDS

#if __has_builtin(__builtin_amdgcn_tanhf)
#define FAST_TANH(x) __builtin_amdgcn_tanhf(x)
#elif __has_builtin(__builtin_amdgcn_tanh_f32)
#define FAST_TANH(x) __builtin_amdgcn_tanh_f32(x)
#else
#define FAST_TANH(x) tanhf(x)
#endif

// ---------------------------------------------------------------- prep kernels
__global__ void k_start(const float* __restrict__ z, const float* __restrict__ c,
                        const float* __restrict__ Ws, const float* __restrict__ bs,
                        float* __restrict__ x) {
    int idx = blockIdx.x * blockDim.x + threadIdx.x;     // B*H = 65536
    if (idx >= B_TOT * HDIM) return;
    int b = idx / HDIM, hh = idx % HDIM;
    const float* wr = Ws + hh * (LATD + 1);
    float s = bs[hh];
#pragma unroll
    for (int k = 0; k < LATD; ++k) s += z[b * LATD + k] * wr[k];
    s += c[b] * wr[LATD];
    x[idx] = s;
}

__global__ void k_xproj(const float* __restrict__ x, const float* __restrict__ Wih,
                        const float* __restrict__ bih, const float* __restrict__ bhh,
                        float* __restrict__ xproj) {
    int idx = blockIdx.x * blockDim.x + threadIdx.x;     // B*4H = 262144
    if (idx >= B_TOT * G4) return;
    int b = idx >> 9, n = idx & 511;
    float s = bih[n] + bhh[n];
    const float* wr = Wih + n * HDIM;
    const float* xr = x + b * HDIM;
#pragma unroll 4
    for (int k = 0; k < HDIM; ++k) s += xr[k] * wr[k];
    xproj[idx] = s;
}

__global__ void k_cvt(const float* __restrict__ w, _Float16* __restrict__ w16) {
    int idx = blockIdx.x * blockDim.x + threadIdx.x;     // 4H*H = 65536
    if (idx < G4 * HDIM) w16[idx] = (_Float16)w[idx];
}

// ---------------------------------------------------------------- main kernel
// grid = 32 workgroups (16 batch rows each), block = 256 threads = 8 waves.
// Wave wv owns N-tiles {wv, wv+8, wv+16, wv+24} = gates {i,f,g,o} for hidden
// columns [wv*16, wv*16+16): gate math + cell state entirely in registers.
// Wave 0 additionally runs the 16x12 output projection on the WMMA pipe.
__global__ __launch_bounds__(256) void lstm_main(
    const float* __restrict__ xproj,        // [B][4H] f32
    const _Float16* __restrict__ whh16g,    // [4H][H] f16
    const float* __restrict__ Wout,         // [OUTD][H] f32
    const float* __restrict__ bout,         // [OUTD]
    const int* __restrict__ seqp,           // seq_len
    float* __restrict__ out)                // [B][T][OUTD]
{
    extern __shared__ char smem[];
    _Float16* whh_s  = (_Float16*)smem;                  // 512*136 halves
    _Float16* h16_s  = whh_s + G4 * WPITCH;              // 16*136 halves
    _Float16* wout_s = h16_s + BT * HPITCH;              // 16*136 halves (rows 12..15 = 0)

    const int tid = threadIdx.x;
    const int bg0 = blockIdx.x * BT;

    // prefetch this block's x_proj rows (gfx1250 global_prefetch path)
    if (tid < BT)
        __builtin_prefetch(&xproj[(size_t)(bg0 + tid) * G4], 0, 1);

    // ---- stage W_hh (f16, padded rows), W_out (f16, 12->16 rows), zero h ----
    for (int ci = tid; ci < G4 * HDIM / 8; ci += 256) {
        int r = ci >> 4, c2 = (ci & 15) << 3;
        *(v8h*)&whh_s[r * WPITCH + c2] = *(const v8h*)&whh16g[r * HDIM + c2];
    }
    for (int i = tid; i < 16 * HDIM; i += 256) {
        int r = i >> 7, c2 = i & 127;
        wout_s[r * WPITCH + c2] = (r < OUTD) ? (_Float16)Wout[r * HDIM + c2]
                                             : (_Float16)0.f;
    }
    for (int i = tid; i < BT * HPITCH; i += 256) h16_s[i] = (_Float16)0.f;

    const int wv    = tid >> 5;
    const int lane  = tid & 31;
    const int mrow  = lane & 15;     // M row (A) / N col within tile (B,C,D)
    const int khalf = lane >> 4;

    // persistent x_proj fragments: xprv[q] is C-operand of tile nt = wv + 8q
    v8f xprv[4];
#pragma unroll
    for (int q = 0; q < 4; ++q) {
        const int col = (wv + 8 * q) * 16 + mrow;
#pragma unroll
        for (int r = 0; r < 8; ++r)
            xprv[q][r] = xproj[(size_t)(bg0 + r + 8 * khalf) * G4 + col];
    }
    float creg[8];                   // persistent cell state
#pragma unroll
    for (int r = 0; r < 8; ++r) creg[r] = 0.f;

    // wave 0: bias C-operand for the output-projection WMMA
    v8f obias;
    {
        float bo = (wv == 0 && mrow < OUTD) ? bout[mrow] : 0.f;
#pragma unroll
        for (int r = 0; r < 8; ++r) obias[r] = bo;
    }

    __syncthreads();

    const int T = seqp[0];
    const _Float16* aBase = &h16_s[mrow * HPITCH + 16 * khalf];
    const _Float16* bBase[4];
#pragma unroll
    for (int q = 0; q < 4; ++q)
        bBase[q] = &whh_s[((wv + 8 * q) * 16 + mrow) * WPITCH + 16 * khalf];
    const _Float16* oBase = &wout_s[mrow * WPITCH + 16 * khalf];

    auto loadA = [&](int kc) -> v16h {
        v8h lo = *(const v8h*)(aBase + kc * 32);
        v8h hi = *(const v8h*)(aBase + kc * 32 + 8);
        return __builtin_shufflevector(lo, hi,
               0,1,2,3,4,5,6,7,8,9,10,11,12,13,14,15);
    };
    auto loadHalves = [&](const _Float16* p) -> v16h {
        v8h lo = *(const v8h*)p;
        v8h hi = *(const v8h*)(p + 8);
        return __builtin_shufflevector(lo, hi,
               0,1,2,3,4,5,6,7,8,9,10,11,12,13,14,15);
    };

    for (int t = 0; t < T; ++t) {
        // ---- gates = x_proj + h @ W_hh^T : 16 WMMAs, 1-deep pipelined LDS ----
        v8f acc[4];
        v16h a_cur = loadA(0);
        v16h b_cur[4];
#pragma unroll
        for (int q = 0; q < 4; ++q) b_cur[q] = loadHalves(bBase[q]);

#pragma unroll
        for (int kc = 0; kc < 4; ++kc) {
            v16h a_nxt = a_cur;
            v16h b_nxt[4];
            if (kc < 3) {
                a_nxt = loadA(kc + 1);
#pragma unroll
                for (int q = 0; q < 4; ++q)
                    b_nxt[q] = loadHalves(bBase[q] + (kc + 1) * 32);
            }
#pragma unroll
            for (int q = 0; q < 4; ++q)
                acc[q] = __builtin_amdgcn_wmma_f32_16x16x32_f16(
                             false, a_cur, false, b_cur[q], (short)0,
                             (kc == 0) ? xprv[q] : acc[q], false, false);
            a_cur = a_nxt;
#pragma unroll
            for (int q = 0; q < 4; ++q) b_cur[q] = b_nxt[q];
        }

        // ---- gate math in registers (acc0=i, acc1=f, acc2=g, acc3=o) ----
        float hv[8];
#pragma unroll
        for (int r = 0; r < 8; ++r) {
            float i_ = 1.f / (1.f + __expf(-acc[0][r]));
            float f_ = 1.f / (1.f + __expf(-acc[1][r]));
            float g_ = FAST_TANH(acc[2][r]);
            float o_ = 1.f / (1.f + __expf(-acc[3][r]));
            float cc = f_ * creg[r] + i_ * g_;
            creg[r] = cc;
            hv[r] = o_ * FAST_TANH(cc);
        }

        __syncthreads();   // all reads of h16(t-1) (gate WMMA + out WMMA) done

        {
            const int n0 = wv * 16 + mrow;
#pragma unroll
            for (int r = 0; r < 8; ++r)
                h16_s[(r + 8 * khalf) * HPITCH + n0] = (_Float16)hv[r];
        }

        __syncthreads();   // h16(t) complete

        // ---- out[b,t,:] = h @ W_out^T + b_out on the WMMA pipe (wave 0) ----
        if (wv == 0) {
            v8f oacc = obias;
#pragma unroll
            for (int kc = 0; kc < 4; ++kc) {
                v16h a = loadA(kc);
                v16h b = loadHalves(oBase + kc * 32);
                oacc = __builtin_amdgcn_wmma_f32_16x16x32_f16(
                           false, a, false, b, (short)0, oacc, false, false);
            }
            if (mrow < OUTD) {
#pragma unroll
                for (int r = 0; r < 8; ++r)
                    out[((size_t)(bg0 + r + 8 * khalf) * T + t) * OUTD + mrow]
                        = oacc[r];
            }
        }
    }
}

// ---------------------------------------------------------------- launch
extern "C" void kernel_launch(void* const* d_in, const int* in_sizes, int n_in,
                              void* d_out, int out_size, void* d_ws, size_t ws_size,
                              hipStream_t stream) {
    const float* z    = (const float*)d_in[0];
    const float* c    = (const float*)d_in[1];
    const float* Ws   = (const float*)d_in[2];
    const float* bs   = (const float*)d_in[3];
    const float* Wih  = (const float*)d_in[4];
    const float* bih  = (const float*)d_in[5];
    const float* Whh  = (const float*)d_in[6];
    const float* bhh  = (const float*)d_in[7];
    const float* Wout = (const float*)d_in[8];
    const float* bout = (const float*)d_in[9];
    const int*   seqp = (const int*)d_in[10];

    float*     xbuf  = (float*)d_ws;                            // [B][H]
    float*     xproj = xbuf + (size_t)B_TOT * HDIM;             // [B][4H]
    _Float16*  whh16 = (_Float16*)(xproj + (size_t)B_TOT * G4); // [4H][H]

    k_start<<<(B_TOT * HDIM + 255) / 256, 256, 0, stream>>>(z, c, Ws, bs, xbuf);
    k_cvt  <<<(G4 * HDIM + 255) / 256, 256, 0, stream>>>(Whh, whh16);
    k_xproj<<<(B_TOT * G4 + 255) / 256, 256, 0, stream>>>(xbuf, Wih, bih, bhh, xproj);

    size_t sh = (size_t)G4 * WPITCH * 2      // W_hh f16
              + (size_t)BT * HPITCH * 2      // h f16
              + (size_t)16 * WPITCH * 2;     // W_out f16 (padded to 16 rows)
    lstm_main<<<B_TOT / BT, 256, sh, stream>>>(xproj, whh16, Wout, bout, seqp,
                                               (float*)d_out);
}
